// AttentionHead_44770739093762
// MI455X (gfx1250) — matrix operand order repaired and checked by reading, not compiled, not run on hardware
//
#include <hip/hip_runtime.h>
#include <hip/hip_bf16.h>

#define MODEL_DIM 1024
#define SEQ       2048
#define BATCH     4
#define VEC       64

#define KSLAB 128          // K-slab staged in LDS per block
#define WROW  (KSLAB + 8)  // +8 halves pad -> spreads banks, keeps 16B align

#if defined(__has_builtin)
#if __has_builtin(__builtin_amdgcn_sched_barrier)
#define SCHED_FENCE() __builtin_amdgcn_sched_barrier(0)
#endif
#endif
#ifndef SCHED_FENCE
#define SCHED_FENCE()
#endif

typedef __attribute__((ext_vector_type(16))) _Float16 v16h;
typedef __attribute__((ext_vector_type(8)))  _Float16 v8h;
typedef __attribute__((ext_vector_type(4)))  _Float16 v4h;
typedef __attribute__((ext_vector_type(8)))  float    v8f;
typedef __attribute__((ext_vector_type(4)))  float    v4f;

__device__ __forceinline__ v8f wmma_f16(v16h a, v16h b, v8f c) {
    // D = A(16x32 f16) * B(32x16 f16) + C(16x16 f32)
    return __builtin_amdgcn_wmma_f32_16x16x32_f16(
        false, a, false, b, (short)0, c, false, false);
}

// A-fragment (16x32, 16-bit): lane holds K-chunks [0..7] and [16..23] (pre-offset by hf*8).
__device__ __forceinline__ v16h a_frag_f32(const float* __restrict__ p) {
    v4f c0 = *(const v4f*)(p + 0);
    v4f c1 = *(const v4f*)(p + 4);
    v4f c2 = *(const v4f*)(p + 16);
    v4f c3 = *(const v4f*)(p + 20);
    v16h f;
#pragma unroll
    for (int i = 0; i < 4; ++i) {
        f[i]      = (_Float16)c0[i];
        f[4 + i]  = (_Float16)c1[i];
        f[8 + i]  = (_Float16)c2[i];
        f[12 + i] = (_Float16)c3[i];
    }
    return f;
}

__device__ __forceinline__ v16h a_frag_h(const _Float16* __restrict__ p) {
    v8h c0 = *(const v8h*)(p);
    v8h c1 = *(const v8h*)(p + 8);
    v16h f;
#pragma unroll
    for (int i = 0; i < 8; ++i) { f[i] = c0[i]; f[8 + i] = c1[i]; }
    return f;
}

// B-fragment (32x16, 16-bit): lane holds K-chunks [0..7] and [16..23] (pre-offset by hf*8).
// 16B-aligned variant (LDS rows are 16B- but not 32B-aligned).
__device__ __forceinline__ v16h b_frag16(const _Float16* p) {
    v8h c0 = *(const v8h*)(p);
    v8h c1 = *(const v8h*)(p + 16);
    v16h f;
#pragma unroll
    for (int i = 0; i < 8; ++i) { f[i] = c0[i]; f[8 + i] = c1[i]; }
    return f;
}

__device__ __forceinline__ v16h b_frag_h(const _Float16* __restrict__ p) {
    return *(const v16h*)p; // 32B contiguous, 32B-aligned
}

// -------- Phase 0: convert W (3 x 64 x 1024) f32 -> f16 once ------------------
__global__ __launch_bounds__(256)
void cvt_w(const float* __restrict__ Wq, const float* __restrict__ Wk,
           const float* __restrict__ Wv, _Float16* __restrict__ Wh) {
    const int i   = blockIdx.x * 256 + threadIdx.x;   // 49152 threads, 4 elems each
    const int mat = i >> 14;                          // 16384 v4-groups per matrix
    const int off = (i & 16383) * 4;
    const float* s = (mat == 0) ? Wq : (mat == 1) ? Wk : Wv;
    v4f v = *(const v4f*)(s + off);
    v4h h;
#pragma unroll
    for (int j = 0; j < 4; ++j) h[j] = (_Float16)v[j];
    *(v4h*)(Wh + (size_t)mat * (VEC * MODEL_DIM) + off) = h;
}

// -------- Phase 1: Q/K/V projections; W staged in LDS, x read exactly once ----
__global__ __launch_bounds__(256)
void proj_qkv(const float* __restrict__ x, const _Float16* __restrict__ Wh,
              _Float16* __restrict__ Qh, _Float16* __restrict__ Kh,
              _Float16* __restrict__ Vt) {
    __shared__ _Float16 lw[3 * VEC * WROW];           // ~52 KB

    const int lane  = threadIdx.x & 31;
    const int wave  = threadIdx.x >> 5;
    const int strip = blockIdx.x * 8 + wave;          // 512 strips of 16 rows
    const int b  = strip >> 7;
    const int s0 = (strip & 127) << 4;
    const int r  = lane & 15;
    const int hf = lane >> 4;

    const float* xp = x + (size_t)(b * SEQ + s0 + r) * MODEL_DIM;

    v8f aq[4], ak[4], av[4];
#pragma unroll
    for (int t = 0; t < 4; ++t) { aq[t] = (v8f){}; ak[t] = (v8f){}; av[t] = (v8f){}; }

    for (int ks = 0; ks < MODEL_DIM; ks += KSLAB) {
        __syncthreads();  // previous slab fully consumed
        // cooperative stage: 3*64 rows x (KSLAB/8) 16B chunks = 3072, exact 12 iters
        for (int idx = threadIdx.x; idx < 3 * VEC * (KSLAB / 8); idx += 256) {
            const int chunk = idx & (KSLAB / 8 - 1);
            const int row   = idx >> 4;               // mat*64 + n
            v8h d = *(const v8h*)(Wh + (size_t)row * MODEL_DIM + ks + chunk * 8);
            *(v8h*)&lw[row * WROW + chunk * 8] = d;
        }
        __syncthreads();

        for (int k0 = 0; k0 < KSLAB; k0 += 32) {
            const int koff = k0 + hf * 8;
            // Issue ALL fragment loads (24x ds_load_b128 + 4x global_load_b128),
            // then fence: no WMMA may be scheduled before the loads are in
            // flight -> single partial wait instead of 12x s_wait_dscnt 0.
            v16h bq[4], bk[4], bv[4];
#pragma unroll
            for (int t = 0; t < 4; ++t) {
                const int n = t * 16 + r;
                bq[t] = b_frag16(&lw[(0 * VEC + n) * WROW + koff]);
                bk[t] = b_frag16(&lw[(1 * VEC + n) * WROW + koff]);
                bv[t] = b_frag16(&lw[(2 * VEC + n) * WROW + koff]);
            }
            v16h a = a_frag_f32(xp + ks + k0 + hf * 8);
            SCHED_FENCE();
#pragma unroll
            for (int t = 0; t < 4; ++t) aq[t] = wmma_f16(a, bq[t], aq[t]);
#pragma unroll
            for (int t = 0; t < 4; ++t) ak[t] = wmma_f16(a, bk[t], ak[t]);
#pragma unroll
            for (int t = 0; t < 4; ++t) av[t] = wmma_f16(a, bv[t], av[t]);
            SCHED_FENCE();
        }
    }
#pragma unroll
    for (int t = 0; t < 4; ++t) {
        const int n = t * 16 + r;
#pragma unroll
        for (int i = 0; i < 8; ++i) {
            const int m = s0 + i + hf * 8;
            Qh[(size_t)(b * SEQ + m) * VEC + n] = (_Float16)aq[t][i];
            Kh[(size_t)(b * SEQ + m) * VEC + n] = (_Float16)ak[t][i];
            Vt[((size_t)b * VEC + n) * SEQ + m] = (_Float16)av[t][i]; // transposed
        }
    }
}

// -------- Phase 2: causal flash attention, 16 queries / wave ------------------
__global__ __launch_bounds__(256)
void attn(const _Float16* __restrict__ Qh, const _Float16* __restrict__ Kh,
          const _Float16* __restrict__ Vt, float* __restrict__ out) {
    const int lane  = threadIdx.x & 31;
    const int wave  = threadIdx.x >> 5;
    const int strip = blockIdx.x * 8 + wave;
    const int b  = strip >> 7;
    const int q0 = (strip & 127) << 4;
    const int r  = lane & 15;
    const int hf = lane >> 4;
    const float scale   = 0.125f; // 1/sqrt(64)
    const float NEG_INF = -__builtin_inff();

    const _Float16* qrow = Qh + (size_t)(b * SEQ + q0 + r) * VEC;
    v16h qb0 = b_frag_h(qrow + hf * 16);        // d 0..31
    v16h qb1 = b_frag_h(qrow + 32 + hf * 16);   // d 32..63

    v8f o[4];
#pragma unroll
    for (int t = 0; t < 4; ++t) o[t] = (v8f){};
    float mrun = NEG_INF, lrun = 0.f;
    const int qg = q0 + r;

    for (int k0 = 0; k0 < q0 + 16; k0 += 32) {   // wave-uniform causal bound
        const _Float16* kp = Kh + (size_t)(b * SEQ + k0 + r) * VEC + hf * 16;
        // All K and V fragment loads in one clause; fence so V latency hides
        // under the score WMMAs + softmax instead of stalling the PV WMMAs.
        v16h ka[4];
        ka[0] = a_frag_h(kp);
        ka[1] = a_frag_h(kp + 32);
        ka[2] = a_frag_h(kp + 16 * VEC);
        ka[3] = a_frag_h(kp + 16 * VEC + 32);
        v16h va[4];
#pragma unroll
        for (int t = 0; t < 4; ++t)
            va[t] = a_frag_h(Vt + ((size_t)b * VEC + t * 16 + r) * SEQ + k0 + hf * 16);
        if (k0 + 32 < q0 + 16) {                 // wave-uniform: prefetch next tiles
            __builtin_prefetch(kp + 32 * VEC, 0, 0);
            __builtin_prefetch(kp + 48 * VEC, 0, 0);
#pragma unroll
            for (int t = 0; t < 4; ++t)
                __builtin_prefetch(
                    Vt + ((size_t)b * VEC + t * 16 + r) * SEQ + (k0 + 32) + hf * 8, 0, 0);
        }
        SCHED_FENCE();

        // S^T = K * Q^T  (two 16-key tiles, K-dim = 64)
        v8f s0 = (v8f){}, s1 = (v8f){};
        s0 = wmma_f16(ka[0], qb0, s0);
        s0 = wmma_f16(ka[1], qb1, s0);
        s1 = wmma_f16(ka[2], qb0, s1);
        s1 = wmma_f16(ka[3], qb1, s1);

        // mask + scale + running max (row q lives in lane r and lane r+16)
        float e0[8], e1[8];
        float mt = NEG_INF;
#pragma unroll
        for (int i = 0; i < 8; ++i) {
            const int kg0 = k0 + i + hf * 8;
            e0[i] = (kg0      <= qg) ? s0[i] * scale : NEG_INF;
            e1[i] = (kg0 + 16 <= qg) ? s1[i] * scale : NEG_INF;
            mt = fmaxf(mt, fmaxf(e0[i], e1[i]));
        }
        mt = fmaxf(mt, __shfl_xor(mt, 16, 32));
        const float mnew  = fmaxf(mrun, mt);
        const float alpha = __expf(mrun - mnew);
        lrun *= alpha;
#pragma unroll
        for (int t = 0; t < 4; ++t)
#pragma unroll
            for (int i = 0; i < 8; ++i) o[t][i] *= alpha;

        float ls = 0.f;
#pragma unroll
        for (int i = 0; i < 8; ++i) {
            e0[i] = (e0[i] == NEG_INF) ? 0.f : __expf(e0[i] - mnew);
            e1[i] = (e1[i] == NEG_INF) ? 0.f : __expf(e1[i] - mnew);
            ls += e0[i] + e1[i];
        }
        lrun += ls + __shfl_xor(ls, 16, 32);
        mrun = mnew;

        // Build P^T B-fragment (K=32 keys): lane hf=0 needs keys 0..15, hf=1 keys 16..31.
        v16h pf;
#pragma unroll
        for (int i = 0; i < 8; ++i) {
            const float p0 = __shfl_xor(e0[i], 16, 32);
            const float p1 = __shfl_xor(e1[i], 16, 32);
            pf[i]     = (_Float16)((hf == 0) ? e0[i] : p1);
            pf[8 + i] = (_Float16)((hf == 0) ? p0    : e1[i]);
        }

        // O^T += V^T * P^T  (4 d-tiles; fragments already in registers)
#pragma unroll
        for (int t = 0; t < 4; ++t) o[t] = wmma_f16(va[t], pf, o[t]);
    }

    const float inv = 1.f / lrun;
#pragma unroll
    for (int t = 0; t < 4; ++t)
#pragma unroll
        for (int i = 0; i < 8; ++i)
            out[(size_t)(b * SEQ + qg) * VEC + t * 16 + i + hf * 8] = o[t][i] * inv;
}

extern "C" void kernel_launch(void* const* d_in, const int* in_sizes, int n_in,
                              void* d_out, int out_size, void* d_ws, size_t ws_size,
                              hipStream_t stream) {
    const float* x  = (const float*)d_in[0];
    const float* Wq = (const float*)d_in[1];
    const float* Wk = (const float*)d_in[2];
    const float* Wv = (const float*)d_in[3];
    float* out = (float*)d_out;

    _Float16* Qh = (_Float16*)d_ws;                        // 1 MB
    _Float16* Kh = Qh + (size_t)BATCH * SEQ * VEC;         // 1 MB
    _Float16* Vt = Kh + (size_t)BATCH * SEQ * VEC;         // 1 MB (transposed [B][64][S])
    _Float16* Wh = Vt + (size_t)BATCH * SEQ * VEC;         // 384 KB ([3][64][1024] f16)

    dim3 blk(256);                                         // 8 waves (wave32)
    cvt_w   <<<dim3(192), blk, 0, stream>>>(Wq, Wk, Wv, Wh);
    proj_qkv<<<dim3(BATCH * SEQ / 16 / 8), blk, 0, stream>>>(x, Wh, Qh, Kh, Vt);
    attn    <<<dim3(BATCH * SEQ / 16 / 8), blk, 0, stream>>>(Qh, Kh, Vt, out);
}
